// Hgru3_42348377539184
// MI455X (gfx1250) — matrix operand order repaired and checked by reading, not compiled, not run on hardware
//
#include <hip/hip_runtime.h>
#include <hip/hip_bf16.h>

typedef __attribute__((ext_vector_type(16))) __bf16 v16bf;
typedef __attribute__((ext_vector_type(8)))  float  v8f;
typedef __attribute__((ext_vector_type(4)))  unsigned int u32x4;
typedef __attribute__((ext_vector_type(8)))  int i32x8;
typedef __attribute__((ext_vector_type(4)))  int i32x4;

#define DB 2
#define DN 4096
#define DE 1024
#define DR 128
#define DH 8
#define DM (DB*DN)   // 8192

#if defined(__has_builtin)
#if __has_builtin(__builtin_amdgcn_tensor_load_to_lds) && __has_builtin(__builtin_amdgcn_s_wait_tensorcnt)
#define HAVE_TDM 1
#endif
#endif
#ifndef HAVE_TDM
#define HAVE_TDM 0
#endif

// ---------- scalar helpers ----------
static __device__ __forceinline__ unsigned short f2bf(float f) {
    union { float f; unsigned u; } c; c.f = f;
    unsigned r = c.u + 0x7FFFu + ((c.u >> 16) & 1u);
    return (unsigned short)(r >> 16);
}
static __device__ __forceinline__ float bf2f(unsigned short b) {
    union { unsigned u; float f; } c; c.u = ((unsigned)b) << 16; return c.f;
}
static __device__ __forceinline__ float sigmf(float x) { return 1.f / (1.f + __expf(-x)); }
static __device__ __forceinline__ float actf(int act, float v) {
    if (act == 1) return v * sigmf(v);   // silu
    if (act == 2) return sigmf(v);       // sigmoid
    return v;
}

#if HAVE_TDM
// LDS byte offset of a generic pointer into shared memory
static __device__ __forceinline__ unsigned lds_off(const void* p) {
    return (unsigned)(__UINTPTR_TYPE__)((const __attribute__((address_space(3))) void*)p);
}
// 2D tile DMA: global (row-major, element=2B) -> LDS, optional LDS row padding.
// dims/strides in 2-byte element units. OOB rows/cols read as zero.
static __device__ __forceinline__ void tdm_load_2d(const void* gptr, const void* lptr,
                                                   unsigned tensor_d0, unsigned tensor_d1,
                                                   unsigned tile_d0, unsigned tile_d1,
                                                   unsigned long long stride0,
                                                   unsigned pad_en, unsigned pad_interval,
                                                   unsigned pad_amount) {
    const unsigned long long ga = (unsigned long long)gptr;
    u32x4 g0;
    g0[0] = 1u;                                            // count=1, user mode
    g0[1] = lds_off(lptr);                                 // lds_addr
    g0[2] = (unsigned)(ga & 0xFFFFFFFFu);                  // global_addr[31:0]
    g0[3] = (unsigned)((ga >> 32) & 0x01FFFFFFu) | 0x80000000u; // addr[56:32] | type=2
    i32x8 g1;
    g1[0] = (int)((1u << 16) | (pad_en << 20) | (pad_interval << 22) | (pad_amount << 25)); // data_size=2B
    g1[1] = (int)((tensor_d0 & 0xFFFFu) << 16);
    g1[2] = (int)((tensor_d0 >> 16) | ((tensor_d1 & 0xFFFFu) << 16));
    g1[3] = (int)((tensor_d1 >> 16) | (tile_d0 << 16));
    g1[4] = (int)(tile_d1 & 0xFFFFu);                      // tile_dim1 (tile_dim2=0)
    g1[5] = (int)(unsigned)(stride0 & 0xFFFFFFFFull);
    g1[6] = (int)(unsigned)((stride0 >> 32) & 0xFFFFull);
    g1[7] = 0;
    const i32x4 z4 = {0, 0, 0, 0};
    const i32x8 z8 = {0, 0, 0, 0, 0, 0, 0, 0};
    __builtin_amdgcn_tensor_load_to_lds(g0, g1, z4, z4, z8, 0);
}
#endif

// ---------- WMMA fragment gathers (wave32, ISA 7.12.2 layouts) ----------
static __device__ __forceinline__ v16bf fragA(const unsigned short* lds, int LD,
                                              int mBase, int kBase, int lane) {
    const int m  = mBase + (lane & 15);
    const int ks = kBase + ((lane >> 4) << 3);
    const unsigned short* p = lds + (size_t)m * LD + ks;
    union { v16bf v; unsigned short u[16]; } f;
#pragma unroll
    for (int e = 0; e < 8; ++e) { f.u[e] = p[e]; f.u[8 + e] = p[16 + e]; }
    return f.v;
}
static __device__ __forceinline__ v16bf fragB(const unsigned short* lds, int LD,
                                              int nBase, int kBase, int lane) {
    const int n  = nBase + (lane & 15);
    const int ks = kBase + ((lane >> 4) << 3);
    const unsigned short* p = lds + (size_t)n * LD + ks;
    union { v16bf v; unsigned short u[16]; } f;
#pragma unroll
    for (int e = 0; e < 8; ++e) { f.u[e] = p[e]; f.u[8 + e] = p[16 + e]; }
    return f.v;
}
static __device__ __forceinline__ v8f wmma_bf16(v16bf a, v16bf b, v8f c) {
    return __builtin_amdgcn_wmma_f32_16x16x32_bf16(false, a, false, b, (short)0, c, false, false);
}

// ---------- f32 -> bf16 convert ----------
__global__ void cvt_f32_bf16(const float* __restrict__ src, unsigned short* __restrict__ dst, size_t n) {
    size_t i = (size_t)blockIdx.x * blockDim.x + threadIdx.x;
    size_t st = (size_t)gridDim.x * blockDim.x;
    for (; i < n; i += st) dst[i] = f2bf(src[i]);
}

// ---------- tiled bf16 WMMA GEMM: out = act(A[M,K] @ W[Nn,K]^T) ----------
__global__ __launch_bounds__(256) void gemm_bf16(const unsigned short* __restrict__ A,
                                                 const unsigned short* __restrict__ W,
                                                 float* __restrict__ outF,
                                                 unsigned short* __restrict__ outB,
                                                 int M, int Nn, int K, int act) {
    const int tid = threadIdx.x, lane = tid & 31, wave = tid >> 5;
    const int m0 = blockIdx.x * 128, n0 = blockIdx.y * 128;
    const int wm = (wave & 1) * 64;       // 2 M-bands of 64
    const int wn = (wave >> 1) * 32;      // 4 N-bands of 32
    v8f acc[4][2] = {};

#if HAVE_TDM
    __shared__ __align__(16) unsigned short As[2][128 * 32];
    __shared__ __align__(16) unsigned short Bs[2][128 * 32];
    const int nk = K / 32;
    // TDM staging: one 128x32 bf16 tile per descriptor; OOB rows (>= Nn) read as zero.
    auto stage = [&](int buf, int k0) {
        if (wave == 0) {
            tdm_load_2d(A + (size_t)m0 * K + k0, &As[buf][0],
                        (unsigned)(K - k0), (unsigned)(M - m0), 32, 128,
                        (unsigned long long)K, 0, 0, 0);
            tdm_load_2d(W + (size_t)n0 * K + k0, &Bs[buf][0],
                        (unsigned)(K - k0), (unsigned)(Nn - n0), 32, 128,
                        (unsigned long long)K, 0, 0, 0);
        }
    };
    stage(0, 0);
    int cur = 0;
    for (int kb = 0; kb < nk; ++kb) {
        if (kb + 1 < nk) stage(cur ^ 1, (kb + 1) * 32);       // prefetch next tile pair
        if (wave == 0) {
            if (kb + 1 < nk) __builtin_amdgcn_s_wait_tensorcnt(2);  // current pair done (in-order)
            else             __builtin_amdgcn_s_wait_tensorcnt(0);
        }
        __syncthreads();
        v16bf bf0 = fragB(&Bs[cur][0], 32, wn + 0, 0, lane);
        v16bf bf1 = fragB(&Bs[cur][0], 32, wn + 16, 0, lane);
#pragma unroll
        for (int mi = 0; mi < 4; ++mi) {
            v16bf af = fragA(&As[cur][0], 32, wm + mi * 16, 0, lane);
            acc[mi][0] = wmma_bf16(af, bf0, acc[mi][0]);
            acc[mi][1] = wmma_bf16(af, bf1, acc[mi][1]);
        }
        __syncthreads();
        cur ^= 1;
    }
#else
    __shared__ __align__(16) unsigned short As[128 * 32];
    __shared__ __align__(16) unsigned short Bs[128 * 32];
    for (int k0 = 0; k0 < K; k0 += 32) {
        __syncthreads();
        {
            const int r = tid >> 1, half = (tid & 1) * 16;
            const unsigned short* src = A + (size_t)(m0 + r) * K + k0 + half;
            *(uint4*)&As[r * 32 + half]     = ((const uint4*)src)[0];
            *(uint4*)&As[r * 32 + half + 8] = ((const uint4*)src)[1];
            uint4 z0 = {0, 0, 0, 0}, z1 = {0, 0, 0, 0};
            if (n0 + r < Nn) {
                const unsigned short* ws = W + (size_t)(n0 + r) * K + k0 + half;
                z0 = ((const uint4*)ws)[0]; z1 = ((const uint4*)ws)[1];
            }
            *(uint4*)&Bs[r * 32 + half]     = z0;
            *(uint4*)&Bs[r * 32 + half + 8] = z1;
        }
        __syncthreads();
        v16bf bf0 = fragB(Bs, 32, wn + 0, 0, lane);
        v16bf bf1 = fragB(Bs, 32, wn + 16, 0, lane);
#pragma unroll
        for (int mi = 0; mi < 4; ++mi) {
            v16bf af = fragA(As, 32, wm + mi * 16, 0, lane);
            acc[mi][0] = wmma_bf16(af, bf0, acc[mi][0]);
            acc[mi][1] = wmma_bf16(af, bf1, acc[mi][1]);
        }
    }
#endif

    const int mlo = (lane >> 4) << 3, nl = lane & 15;
#pragma unroll
    for (int mi = 0; mi < 4; ++mi)
#pragma unroll
        for (int ni = 0; ni < 2; ++ni)
#pragma unroll
            for (int r = 0; r < 8; ++r) {
                const int m = m0 + wm + mi * 16 + r + mlo;
                const int n = n0 + wn + ni * 16 + nl;
                if (n < Nn) {
                    float v = actf(act, acc[mi][ni][r]);
                    if (outF) outF[(size_t)m * Nn + n] = v;
                    if (outB) outB[(size_t)m * Nn + n] = f2bf(v);
                }
            }
}

// ---------- decay gate ----------
__global__ void decay_kernel(const float* __restrict__ F, const float* __restrict__ llb,
                             float* __restrict__ logf) {
    int idx = blockIdx.x * blockDim.x + threadIdx.x;
    if (idx >= DB * DN * DH) return;
    const int h = idx % DH, m = idx / DH, b = m / DN, n = m % DN;
    const float lb = __expf(llb[h]);
    const float fg = lb + (1.f - lb) * sigmf(F[(size_t)m * DH + h]);
    logf[((size_t)(b * DH + h)) * DN + n] = __logf(fg);
}

// ---------- Householder ----------
__global__ __launch_bounds__(256) void householder_kernel(const float* __restrict__ Q,
                                                          const float* __restrict__ beta,
                                                          unsigned short* __restrict__ qh) {
    const int m = blockIdx.x, tid = threadIdx.x;
    __shared__ float rs[256], rd[256];
    const float* q = Q + (size_t)m * DE;
    const float* be = beta + (size_t)m * DE;
    float ss = 0.f, dot = 0.f;
    for (int e = tid; e < DE; e += 256) { float b = be[e]; ss += b * b; dot += q[e] * b; }
    rs[tid] = ss; rd[tid] = dot; __syncthreads();
    for (int s = 128; s > 0; s >>= 1) {
        if (tid < s) { rs[tid] += rs[tid + s]; rd[tid] += rd[tid + s]; }
        __syncthreads();
    }
    const float coef = 2.f * rd[0] / (rs[0] + 1e-12f);
    for (int e = tid; e < DE; e += 256)
        qh[(size_t)m * DE + e] = f2bf(q[e] - coef * be[e]);
}

// ---------- chunked gated linear attention (one WG per (b,h)) ----------
#define GC 64
__global__ __launch_bounds__(256) void gla_kernel(const unsigned short* __restrict__ qh,
                                                  const unsigned short* __restrict__ kh,
                                                  const unsigned short* __restrict__ vh,
                                                  const float* __restrict__ logf,
                                                  float* __restrict__ o) {
    const int bh = blockIdx.x, b = bh / DH, h = bh % DH;
    const int tid = threadIdx.x, lane = tid & 31, wave = tid >> 5;
    __shared__ __align__(16) unsigned short Qs[GC * 136];    // [i][r], padded rows
    __shared__ __align__(16) unsigned short Ks[GC * 136];    // [j][r]
    __shared__ __align__(16) unsigned short VT[DR * 72];     // [v][j]
    __shared__ __align__(16) unsigned short KdT[DR * 72];    // [r][j]
    __shared__ __align__(16) unsigned short Pb[GC * 72];     // [i][j]
    __shared__ __align__(16) unsigned short ST[DR * 136];    // state transposed [v][r]
    __shared__ float aa[GC + 1];

    for (int i = tid; i < DR * 136; i += 256) ST[i] = 0;
    const float scale = rsqrtf((float)DR);
    const float* gptr = logf + (size_t)bh * DN;

    for (int c0 = 0; c0 < DN; c0 += GC) {
        __syncthreads();
        const size_t gbase = ((size_t)(b * DN + c0)) * DE + h * DR;
#if HAVE_TDM
        // 64 rows x 128 elems, global row stride DE; pad LDS rows by 4 DWORDs every
        // 64 DWORDs -> 136-ushort LDS rows (pad_interval code 5 = 64 DW, amount code 3 = 4 DW)
        if (wave == 0) {
            tdm_load_2d(qh + gbase, Qs, 128, (unsigned)DN, 128, GC,
                        (unsigned long long)DE, 1, 5, 3);
            tdm_load_2d(kh + gbase, Ks, 128, (unsigned)DN, 128, GC,
                        (unsigned long long)DE, 1, 5, 3);
        }
#else
        for (int i = tid; i < GC * 8; i += 256) {
            const int r = i >> 3, seg = (i & 7) * 16;
            const size_t g = gbase + (size_t)r * DE + seg;
            *(uint4*)&Qs[r * 136 + seg]     = ((const uint4*)&qh[g])[0];
            *(uint4*)&Qs[r * 136 + seg + 8] = ((const uint4*)&qh[g])[1];
            *(uint4*)&Ks[r * 136 + seg]     = ((const uint4*)&kh[g])[0];
            *(uint4*)&Ks[r * 136 + seg + 8] = ((const uint4*)&kh[g])[1];
        }
#endif
        // V transposed (manual scatter; TDM has no transpose)
        for (int i = tid; i < GC * 16; i += 256) {
            const int r = i >> 4, seg = (i & 15) * 8;
            const uint4 d = *(const uint4*)&vh[gbase + (size_t)r * DE + seg];
            const unsigned short* ds = (const unsigned short*)&d;
#pragma unroll
            for (int e = 0; e < 8; ++e) VT[(size_t)(seg + e) * 72 + r] = ds[e];
        }
        if (tid == 0) {
            float s = 0.f;
            for (int i = 0; i < GC; ++i) { s += gptr[c0 + i]; aa[i] = s; }
        }
#if HAVE_TDM
        if (wave == 0) __builtin_amdgcn_s_wait_tensorcnt(0);
#endif
        __syncthreads();
        const float aC = aa[GC - 1];

        // P = Q K^T, masked with scale*exp(a_i - a_j), j<=i
        for (int t = wave * 2; t < wave * 2 + 2; ++t) {
            const int tm = t >> 2, tn = t & 3;
            v8f acc = {};
            for (int k0 = 0; k0 < DR; k0 += 32)
                acc = wmma_bf16(fragA(Qs, 136, tm * 16, k0, lane),
                                fragB(Ks, 136, tn * 16, k0, lane), acc);
            const int mlo = (lane >> 4) << 3, nl = lane & 15;
#pragma unroll
            for (int r = 0; r < 8; ++r) {
                const int i = tm * 16 + r + mlo, j = tn * 16 + nl;
                const float v = (j <= i) ? acc[r] * scale * __expf(aa[i] - aa[j]) : 0.f;
                Pb[i * 72 + j] = f2bf(v);
            }
        }
        __syncthreads();
        // scale Q in place; KdT[r][j] = k_j[r] * e^{aC-a_j}
        for (int i = tid; i < GC * DR; i += 256) {
            const int r = i >> 7, cc = i & 127;
            Qs[r * 136 + cc] = f2bf(bf2f(Qs[r * 136 + cc]) * scale * __expf(aa[r]));
            KdT[cc * 72 + r] = f2bf(bf2f(Ks[r * 136 + cc]) * __expf(aC - aa[r]));
        }
        __syncthreads();
        // O = P@V + Qs@S
        for (int t = wave * 4; t < wave * 4 + 4; ++t) {
            const int tm = t >> 3, tn = t & 7;
            v8f acc = {};
            for (int k0 = 0; k0 < GC; k0 += 32)
                acc = wmma_bf16(fragA(Pb, 72, tm * 16, k0, lane),
                                fragB(VT, 72, tn * 16, k0, lane), acc);
            for (int k0 = 0; k0 < DR; k0 += 32)
                acc = wmma_bf16(fragA(Qs, 136, tm * 16, k0, lane),
                                fragB(ST, 136, tn * 16, k0, lane), acc);
            const int mlo = (lane >> 4) << 3, nl = lane & 15;
#pragma unroll
            for (int r = 0; r < 8; ++r) {
                const int i = tm * 16 + r + mlo, v = tn * 16 + nl;
                o[((size_t)(b * DN + c0 + i)) * DE + h * DR + v] = acc[r];
            }
        }
        __syncthreads();
        // S <- e^{aC} S + V^T Kd   (transposed state)
        const float dec = __expf(aC);
        for (int tr = 0; tr < 8; ++tr) {
            v8f acc = {};
            for (int k0 = 0; k0 < GC; k0 += 32)
                acc = wmma_bf16(fragA(VT, 72, wave * 16, k0, lane),
                                fragB(KdT, 72, tr * 16, k0, lane), acc);
            const int mlo = (lane >> 4) << 3, nl = lane & 15;
#pragma unroll
            for (int r = 0; r < 8; ++r) {
                const int v = wave * 16 + r + mlo, rr = tr * 16 + nl;
                ST[v * 136 + rr] = f2bf(dec * bf2f(ST[v * 136 + rr]) + acc[r]);
            }
        }
    }
}

// ---------- gated LayerNorm -> bf16 ----------
__global__ __launch_bounds__(256) void ln_gate_kernel(const float* __restrict__ o,
                                                      const float* __restrict__ gate,
                                                      const float* __restrict__ gamma,
                                                      unsigned short* __restrict__ y) {
    const int m = blockIdx.x, tid = threadIdx.x;
    __shared__ float rs[256], rq[256];
    float gv[4], sum = 0.f, sq = 0.f;
#pragma unroll
    for (int j = 0; j < 4; ++j) {
        const int e = tid + j * 256;
        const float g = o[(size_t)m * DE + e] * gate[(size_t)m * DE + e];
        gv[j] = g; sum += g; sq += g * g;
    }
    rs[tid] = sum; rq[tid] = sq; __syncthreads();
    for (int s = 128; s > 0; s >>= 1) {
        if (tid < s) { rs[tid] += rs[tid + s]; rq[tid] += rq[tid + s]; }
        __syncthreads();
    }
    const float mu = rs[0] * (1.f / DE);
    const float var = rq[0] * (1.f / DE) - mu * mu;
    const float inv = rsqrtf(var + 1e-5f);
#pragma unroll
    for (int j = 0; j < 4; ++j) {
        const int e = tid + j * 256;
        y[(size_t)m * DE + e] = f2bf((gv[j] - mu) * inv * gamma[e]);
    }
}

// ---------- host ----------
extern "C" void kernel_launch(void* const* d_in, const int* in_sizes, int n_in,
                              void* d_out, int out_size, void* d_ws, size_t ws_size,
                              hipStream_t stream) {
    const float* x    = (const float*)d_in[0];
    const float* llb  = (const float*)d_in[1];
    const float* Wq   = (const float*)d_in[2];
    const float* Wk   = (const float*)d_in[3];
    const float* Wv   = (const float*)d_in[4];
    const float* Wf   = (const float*)d_in[5];
    const float* Wb1  = (const float*)d_in[6];
    const float* Wb2  = (const float*)d_in[7];
    const float* Wg1  = (const float*)d_in[8];
    const float* Wg2  = (const float*)d_in[9];
    const float* gamma= (const float*)d_in[10];
    const float* Wo   = (const float*)d_in[11];
    float* out = (float*)d_out;

    char* ws = (char*)d_ws; size_t off = 0;
    auto alloc = [&](size_t bytes) -> void* {
        void* p = ws + off; off = (off + bytes + 255) & ~(size_t)255; return p;
    };
    typedef unsigned short us;
    us* xb   = (us*)alloc((size_t)DM * DE * 2);
    us* wqb  = (us*)alloc((size_t)DE * DE * 2);
    us* wkb  = (us*)alloc((size_t)DE * DE * 2);
    us* wvb  = (us*)alloc((size_t)DE * DE * 2);
    us* wob  = (us*)alloc((size_t)DE * DE * 2);
    us* wfb  = (us*)alloc((size_t)DH * DE * 2);
    us* wb1b = (us*)alloc((size_t)DR * DE * 2);
    us* wb2b = (us*)alloc((size_t)DE * DR * 2);
    us* wg1b = (us*)alloc((size_t)DR * DE * 2);
    us* wg2b = (us*)alloc((size_t)DE * DR * 2);
    float* Qf   = (float*)alloc((size_t)DM * DE * 4);
    us*    Kb   = (us*)alloc((size_t)DM * DE * 2);
    us*    Vb   = (us*)alloc((size_t)DM * DE * 2);
    float* Ff   = (float*)alloc((size_t)DM * DH * 4);
    us*    xb1  = (us*)alloc((size_t)DM * DR * 2);
    us*    xg1  = (us*)alloc((size_t)DM * DR * 2);
    float* betaF= (float*)alloc((size_t)DM * DE * 4);
    float* gateF= (float*)alloc((size_t)DM * DE * 4);
    us*    qhb  = (us*)alloc((size_t)DM * DE * 2);
    float* logf = (float*)alloc((size_t)DB * DH * DN * 4);
    float* oF   = (float*)alloc((size_t)DM * DE * 4);
    us*    yb   = (us*)alloc((size_t)DM * DE * 2);
    (void)ws_size; (void)n_in; (void)in_sizes; (void)out_size;

    auto cvt = [&](const float* s, us* d, size_t n) {
        int g = (int)((n + 255) / 256); if (g > 4096) g = 4096;
        cvt_f32_bf16<<<g, 256, 0, stream>>>(s, d, n);
    };
    cvt(x,  xb,  (size_t)DM * DE);
    cvt(Wq, wqb, (size_t)DE * DE);  cvt(Wk, wkb, (size_t)DE * DE);
    cvt(Wv, wvb, (size_t)DE * DE);  cvt(Wo, wob, (size_t)DE * DE);
    cvt(Wf, wfb, (size_t)DH * DE);
    cvt(Wb1, wb1b, (size_t)DR * DE); cvt(Wb2, wb2b, (size_t)DE * DR);
    cvt(Wg1, wg1b, (size_t)DR * DE); cvt(Wg2, wg2b, (size_t)DE * DR);

    auto gemm = [&](const us* A, const us* W, float* oF_, us* oB_, int M, int Nn, int K, int act) {
        dim3 grid(M / 128, (Nn + 127) / 128);
        gemm_bf16<<<grid, 256, 0, stream>>>(A, W, oF_, oB_, M, Nn, K, act);
    };
    gemm(xb, wqb,  Qf,   nullptr, DM, DE, DE, 1);   // q = silu(x Wq^T), f32
    gemm(xb, wkb,  nullptr, Kb,   DM, DE, DE, 1);   // k = silu(x Wk^T), bf16
    gemm(xb, wvb,  nullptr, Vb,   DM, DE, DE, 0);   // v, bf16
    gemm(xb, wfb,  Ff,   nullptr, DM, DH, DE, 0);   // f logits
    gemm(xb, wb1b, nullptr, xb1,  DM, DR, DE, 0);   // x Wb1^T
    gemm(xb, wg1b, nullptr, xg1,  DM, DR, DE, 0);   // x Wg1^T
    gemm(xb1, wb2b, betaF, nullptr, DM, DE, DR, 1); // beta = silu(.)
    gemm(xg1, wg2b, gateF, nullptr, DM, DE, DR, 2); // gate = sigmoid(.)

    decay_kernel<<<(DB * DN * DH + 255) / 256, 256, 0, stream>>>(Ff, llb, logf);
    householder_kernel<<<DM, 256, 0, stream>>>(Qf, betaF, qhb);
    gla_kernel<<<DB * DH, 256, 0, stream>>>(qhb, Kb, Vb, logf, oF);
    ln_gate_kernel<<<DM, 256, 0, stream>>>(oF, gateF, gamma, yb);
    gemm(yb, wob, out, nullptr, DM, DE, DE, 0);     // final projection
}
